// Feature_69045894250871
// MI455X (gfx1250) — matrix operand order repaired and checked by reading, not compiled, not run on hardware
//
#include <hip/hip_runtime.h>
#include <hip/hip_bf16.h>

#ifndef __has_builtin
#define __has_builtin(x) 0
#endif
#ifndef __has_include
#define __has_include(x) 0
#endif

typedef _Float16 h16 __attribute__((ext_vector_type(16)));
typedef _Float16 h8  __attribute__((ext_vector_type(8)));
typedef _Float16 h2  __attribute__((ext_vector_type(2)));
typedef float    f8  __attribute__((ext_vector_type(8)));
typedef unsigned int u32x4 __attribute__((ext_vector_type(4)));
typedef int          i32x4 __attribute__((ext_vector_type(4)));
typedef int          i32x8 __attribute__((ext_vector_type(8)));

// exact element type the async-LDS builtins expect (per hipcc diagnostic):
// '__attribute__((__vector_size__(4 * sizeof(int)))) int'
typedef int vint4 __attribute__((__vector_size__(16)));
typedef __attribute__((address_space(1))) vint4 gvint4;
typedef __attribute__((address_space(3))) vint4 lvint4;

#define NBATCH 64
#define NCAT   128
#define NK     127
#define D1     25
#define D2     50
#define D3     100
#define MB     16   // batch tile == WMMA M

// ---- CDNA5 feature detection (compile-probe driven, always with fallback) --
#if __has_builtin(__builtin_amdgcn_global_load_async_to_lds_b128) && \
    __has_builtin(__builtin_amdgcn_s_wait_asynccnt)
#define HAVE_ASYNC_LDS 1
#else
#define HAVE_ASYNC_LDS 0
#endif

#if __has_include(<hip/amd_detail/amd_gfx1250_TDM.h>)
#define TDM_ARITY 6            // amdgpu-toolchain / therock headers -> 6-arg builtin
#elif __has_builtin(__builtin_amdgcn_tensor_load_to_lds)
#define TDM_ARITY 5            // ROCm 7.2 clang-22 -> 5-arg builtin
#else
#define TDM_ARITY 0
#endif

__device__ __forceinline__ float fast_tanh(float x) {
#if __has_builtin(__builtin_amdgcn_tanhf)
  return __builtin_amdgcn_tanhf(x);        // v_tanh_f32 (TRANS, co-exec w/ WMMA)
#elif __has_builtin(__builtin_amdgcn_tanh_f32)
  return __builtin_amdgcn_tanh_f32(x);
#else
  return tanhf(x);
#endif
}

// generic -> LDS addrspace(3): low 32 bits of a generic LDS address are the
// LDS byte offset (ISA 10.2 aperture rules), so an integer cast is exact.
__device__ __forceinline__ lvint4* to_lds4(void* p) {
  return (lvint4*)(unsigned int)(unsigned long long)p;
}
__device__ __forceinline__ gvint4* to_glob4(const void* p) {
  return (gvint4*)(unsigned long long)p;
}

// cooperative 16B-granule global->LDS stage (async path when available)
__device__ __forceinline__ void stage16(const void* g, void* l, int n16, int tid) {
#if HAVE_ASYNC_LDS
  const char* gp = (const char*)g;
  char* lp = (char*)l;
  for (int i = tid; i < n16; i += 128)
    __builtin_amdgcn_global_load_async_to_lds_b128(
        to_glob4(gp + 16 * (size_t)i), to_lds4(lp + 16 * (size_t)i), 0, 0);
#else
  const uint4* s = (const uint4*)g;
  uint4* d = (uint4*)l;
  for (int i = tid; i < n16; i += 128) d[i] = s[i];
#endif
}

// ---------------------------------------------------------------------------
// Pack W2 (36,50,25) and W3 (36,100,50) f32 -> f16, zero-padded to
// W2h[36][64][32] and W3h[36][112][64] (row-major [o][i]); a WMMA B-fragment
// for N-tile t / lane half h is then two 8-half runs of one row (k = 8h.., 16+8h..).
// ---------------------------------------------------------------------------
__global__ void pack_weights(const float* __restrict__ W2,
                             const float* __restrict__ W3,
                             _Float16* __restrict__ W2h,
                             _Float16* __restrict__ W3h) {
  int ch = blockIdx.x; // 36 channels
  for (int idx = threadIdx.x; idx < 64 * 32; idx += blockDim.x) {
    int o = idx >> 5, i = idx & 31;
    float v = (o < D2 && i < D1) ? W2[(ch * D2 + o) * D1 + i] : 0.f;
    W2h[ch * 2048 + idx] = (_Float16)v;
  }
  for (int idx = threadIdx.x; idx < 112 * 64; idx += blockDim.x) {
    int o = idx >> 6, i = idx & 63;
    float v = (o < D3 && i < D2) ? W3[(ch * D3 + o) * D2 + i] : 0.f;
    W3h[ch * 7168 + idx] = (_Float16)v;
  }
}

// ---------------------------------------------------------------------------
// Fused descriptor kernel. Block = 128 threads (4 wave32), grid = (128, 4).
// ---------------------------------------------------------------------------
__global__ __launch_bounds__(128, 1)
void se_fused(const float* __restrict__ coords,
              const int*   __restrict__ types,
              const float* __restrict__ W1,
              const float* __restrict__ b1,
              const float* __restrict__ b2,
              const float* __restrict__ b3,
              const _Float16* __restrict__ W2h,
              const _Float16* __restrict__ W3h,
              float* __restrict__ out) {
  extern __shared__ char smem[];
  float*    coordsS = (float*)(smem);              // [16][128][3]   24576 B
  _Float16* w3hS    = (_Float16*)(smem + 24576);   // [6][112][64]   86016 B
  _Float16* w2hS    = (_Float16*)(smem + 110592);  // [6][64][32]    24576 B
  float*    b2S     = (float*)(smem + 135168);     // [6][64]         1536 B
  float*    b3S     = (float*)(smem + 136704);     // [6][112]        2688 B
  float*    w1S     = (float*)(smem + 139392);     // [6][32]          768 B
  float*    b1S     = (float*)(smem + 140160);     // [6][32]          768 B
  float*    aS      = (float*)(smem + 140928);     // [16][4]          256 B
  float*    t1S     = (float*)(smem + 141184);     // [16][4][4]      1024 B
  _Float16* y2S     = (_Float16*)(smem + 142208);  // [16][64]        2048 B
  _Float16* y1S     = (_Float16*)(smem + 144256);  // [16][32]        1024 B
  int*      chS     = (int*)(smem + 145280);       // [128]            512 B
                                                   // total         145792 B
  const int n   = blockIdx.x;
  const int b0  = blockIdx.y * MB;
  const int tid = threadIdx.x;
  const int w   = tid >> 5;
  const int l   = tid & 31;
  const int hi  = l >> 4;   // lane half: selects K-half of A/B fragments
  const int b16 = l & 15;

  const int myType = types[n];

  // ---- stage coords + this atom-type's 6 weight channels into LDS ----
#if TDM_ARITY
  // One TDM DMA for the big 84KB W3 block: 1-D tensor of 10752 qwords.
  if (w == 0) {
    unsigned long long ga = (unsigned long long)(W3h + (size_t)myType * 6 * 7168);
    unsigned lds = (unsigned)(unsigned long long)(void*)w3hS;
    u32x4 g0;
    g0[0] = 1u;                                   // count=1, user mode
    g0[1] = lds;                                  // lds_addr
    g0[2] = (unsigned)ga;                         // global_addr[31:0]
    g0[3] = (unsigned)(ga >> 32) | 0x80000000u;   // global_addr[56:32] | type=2
    i32x8 g1;
    g1[0] = 0x30000;                  // workgroup_mask=0, data_size=3 (8B)
    g1[1] = (int)(10752u << 16);      // tensor_dim0[15:0] at bits[63:48]
    g1[2] = 0;                        // tensor_dim0 hi16 = 0, tensor_dim1 = 0
    g1[3] = (int)(10752u << 16);      // tile_dim0 at bits[127:112]
    g1[4] = 0;                        // tile_dim1 = tile_dim2 = 0 (1-D)
    g1[5] = 10752;                    // tensor_dim0_stride[31:0]
    g1[6] = 0;
    g1[7] = 0;
    i32x4 gz = {0, 0, 0, 0};
#if TDM_ARITY == 5
    __builtin_amdgcn_tensor_load_to_lds(g0, g1, gz, gz, 0);
#else
    i32x8 gz8 = {0, 0, 0, 0, 0, 0, 0, 0};
    __builtin_amdgcn_tensor_load_to_lds(g0, g1, gz, gz, gz8, 0);
#endif
  }
#else
  stage16(W3h + (size_t)myType * 6 * 7168, w3hS, (6 * 7168 * 2) / 16, tid);
#endif
  stage16(coords + (size_t)b0 * NCAT * 3, coordsS, (MB * NCAT * 3 * 4) / 16, tid);
  stage16(W2h + (size_t)myType * 6 * 2048, w2hS, (6 * 2048 * 2) / 16, tid);

  for (int i = tid; i < 6 * 32; i += 128) {
    int c = i >> 5, ii = i & 31;
    int g = myType * 6 + c;
    w1S[i] = (ii < D1) ? W1[g * D1 + ii] : 0.f;
    b1S[i] = (ii < D1) ? b1[g * D1 + ii] : 0.f;
  }
  for (int i = tid; i < 6 * 64; i += 128) {
    int c = i >> 6, o = i & 63;
    b2S[i] = (o < D2) ? b2[(myType * 6 + c) * D2 + o] : 0.f;
  }
  for (int i = tid; i < 6 * 112; i += 128) {
    int c = i / 112, o = i % 112;
    b3S[i] = (o < D3) ? b3[(myType * 6 + c) * D3 + o] : 0.f;
  }
  if (tid < NCAT) chS[tid] = types[tid];

#if HAVE_ASYNC_LDS
  __builtin_amdgcn_s_wait_asynccnt(0);
#endif
#if TDM_ARITY
#if __has_builtin(__builtin_amdgcn_s_wait_tensorcnt)
  __builtin_amdgcn_s_wait_tensorcnt(0);
#endif
#endif
  __syncthreads();

  const float cx = coordsS[(b16 * NCAT + n) * 3 + 0];
  const float cy = coordsS[(b16 * NCAT + n) * 3 + 1];
  const float cz = coordsS[(b16 * NCAT + n) * 3 + 2];

  const int nt0 = 2 * w;               // layer-3 N-tiles: {0,1},{2,3},{4,5},{6}
  const int ntc = (w == 3) ? 1 : 2;

  float G[2][8][3];
  #pragma unroll
  for (int a = 0; a < 2; ++a)
    #pragma unroll
    for (int r = 0; r < 8; ++r)
      G[a][r][0] = G[a][r][1] = G[a][r][2] = 0.f;

  for (int k = 0; k < NK; ++k) {
    __syncthreads();                       // prev-iter LDS reads complete
    const int j = k + (k >= n);
    const int c = chS[j];

    // geometry for batch b = b16 (this lane's A-fragment row)
    float dxv = cx - coordsS[(b16 * NCAT + j) * 3 + 0];
    float dyv = cy - coordsS[(b16 * NCAT + j) * 3 + 1];
    float dzv = cz - coordsS[(b16 * NCAT + j) * 3 + 2];
    float d2   = dxv * dxv + dyv * dyv + dzv * dzv;
    float dinv = 1.f / fmaxf(sqrtf(d2), 1e-12f);
    float s2   = dinv * dinv;
    float ax = dxv * s2, ay = dyv * s2, az = dzv * s2;
    float x = dinv;                        // loc_env_r

    // ---- layer 1 (K=1, VALU+TRANS) directly in WMMA A-fragment layout ----
    h16 a1;
    #pragma unroll
    for (int m = 0; m < 16; ++m) {
      int p = m >> 1, e = m & 1;
      int i = (p < 4) ? (8 * hi + 2 * p + e) : (16 + 8 * hi + 2 * (p - 4) + e);
      float y = 0.f;
      if (i < D1) y = fast_tanh(w1S[c * 32 + i] * x + b1S[c * 32 + i]);
      a1[m] = (_Float16)y;
    }
    if (w == 0) {                          // stage A + Y1 for C-layout consumers
      if (hi == 0) { float4 t = make_float4(ax, ay, az, x);
                     *(float4*)(aS + b16 * 4) = t; }
      #pragma unroll
      for (int p = 0; p < 8; ++p) {
        int i0 = (p < 4) ? (8 * hi + 2 * p) : (16 + 8 * hi + 2 * (p - 4));
        h2 pr; pr[0] = a1[2 * p]; pr[1] = a1[2 * p + 1];
        *(h2*)(y1S + b16 * 32 + i0) = pr;
      }
    }

    // ---- layer 2: 16x64 = 4 N-tiles, K=32 (one WMMA each) ----
    f8 acc2[4];
    #pragma unroll
    for (int t = 0; t < 4; ++t) {
      const _Float16* bp = w2hS + ((c * 64) + (t * 16 + b16)) * 32 + 8 * hi;
      h16 bf = __builtin_shufflevector(*(const h8*)bp, *(const h8*)(bp + 16), 0, 1, 2, 3, 4, 5, 6, 7, 8, 9, 10, 11, 12, 13, 14, 15);
      f8 z = {};
      acc2[t] = __builtin_amdgcn_wmma_f32_16x16x32_f16(
          false, a1, false, bf, (short)0, z, false, false);
    }
    __syncthreads();                       // aS / y1S visible

    // layer-2 epilogue: tanh + concat residual, emit Y2 as f16 in LDS
    #pragma unroll
    for (int t = 0; t < 4; ++t) {
      int o = t * 16 + b16;
      float bias = b2S[c * 64 + o];
      int i2 = o % D1;
      #pragma unroll
      for (int r = 0; r < 8; ++r) {
        int b = r + 8 * hi;
        float v = fast_tanh(acc2[t][r] + bias) + (float)y1S[b * 32 + i2];
        y2S[b * 64 + o] = (_Float16)((o < D2) ? v : 0.f);  // zero K-padding
      }
    }
    __syncthreads();                       // y2S visible

    // env vectors for this lane's C-fragment rows
    float4 av[8];
    #pragma unroll
    for (int r = 0; r < 8; ++r) av[r] = *(float4*)(aS + (r + 8 * hi) * 4);

    // rebuild Y2 as layer-3 A-fragments (two K-steps of 32)
    h16 a30, a31;
    #pragma unroll
    for (int s = 0; s < 2; ++s) {
      h8 lo = *(const h8*)(y2S + b16 * 64 + 32 * s + 8 * hi);
      h8 hb = *(const h8*)(y2S + b16 * 64 + 32 * s + 16 + 8 * hi);
      h16 f;
      #pragma unroll
      for (int m = 0; m < 8; ++m) { f[m] = lo[m]; f[8 + m] = hb[m]; }
      if (s == 0) a30 = f; else a31 = f;
    }

    // ---- layer 3: wave's N-tiles, K=64 (2 chained WMMAs/tile) + G accum ----
    for (int tt = 0; tt < ntc; ++tt) {
      int t = nt0 + tt;
      int o = t * 16 + b16;
      const _Float16* bp = w3hS + (c * 112 + o) * 64;
      h16 bf0 = __builtin_shufflevector(*(const h8*)(bp + 8 * hi), *(const h8*)(bp + 16 + 8 * hi), 0, 1, 2, 3, 4, 5, 6, 7, 8, 9, 10, 11, 12, 13, 14, 15);
      h16 bf1 = __builtin_shufflevector(*(const h8*)(bp + 32 + 8 * hi), *(const h8*)(bp + 48 + 8 * hi), 0, 1, 2, 3, 4, 5, 6, 7, 8, 9, 10, 11, 12, 13, 14, 15);
      f8 z = {};
      f8 acc = __builtin_amdgcn_wmma_f32_16x16x32_f16(
          false, a30, false, bf0, (short)0, z, false, false);
      acc = __builtin_amdgcn_wmma_f32_16x16x32_f16(
          false, a31, false, bf1, (short)0, acc, false, false);
      float bias = b3S[c * 112 + o];
      int i2 = o % D2;
      #pragma unroll
      for (int r = 0; r < 8; ++r) {
        int b = r + 8 * hi;
        float ov = fast_tanh(acc[r] + bias) + (float)y2S[b * 64 + i2];
        G[tt][r][0] += ov * av[r].x;
        G[tt][r][1] += ov * av[r].y;
        G[tt][r][2] += ov * av[r].z;
      }
    }
  } // k loop

  // ---- res[g,f] = sum_a G[g,a] * G[f,a]  (t1 = G[:4,:]) ----
  __syncthreads();
  if (w == 0 && b16 < 4) {                 // tile 0 holds g = 0..15 -> f = b16
    #pragma unroll
    for (int r = 0; r < 8; ++r) {
      int b = r + 8 * hi;
      float4 t = make_float4(G[0][r][0], G[0][r][1], G[0][r][2], 0.f);
      *(float4*)(t1S + b * 16 + b16 * 4) = t;
    }
  }
  __syncthreads();

  for (int tt = 0; tt < ntc; ++tt) {
    int t = nt0 + tt;
    int o = t * 16 + b16;
    if (o >= D3) continue;
    #pragma unroll
    for (int r = 0; r < 8; ++r) {
      int b = r + 8 * hi;
      const float* tb = t1S + b * 16;
      float g0 = G[tt][r][0], g1 = G[tt][r][1], g2 = G[tt][r][2];
      float4 rv;
      rv.x = g0 * tb[0]  + g1 * tb[1]  + g2 * tb[2];
      rv.y = g0 * tb[4]  + g1 * tb[5]  + g2 * tb[6];
      rv.z = g0 * tb[8]  + g1 * tb[9]  + g2 * tb[10];
      rv.w = g0 * tb[12] + g1 * tb[13] + g2 * tb[14];
      size_t off = (((size_t)(b0 + b) * NCAT + n) * D3 + o) * 4;
      *(float4*)(out + off) = rv;
    }
  }
}

extern "C" void kernel_launch(void* const* d_in, const int* in_sizes, int n_in,
                              void* d_out, int out_size, void* d_ws, size_t ws_size,
                              hipStream_t stream) {
  (void)in_sizes; (void)n_in; (void)out_size; (void)ws_size;
  const float* coords = (const float*)d_in[0];
  const int*   typesP = (const int*)d_in[1];
  const float* W1 = (const float*)d_in[2];
  const float* b1 = (const float*)d_in[3];
  const float* W2 = (const float*)d_in[4];
  const float* b2 = (const float*)d_in[5];
  const float* W3 = (const float*)d_in[6];
  const float* b3 = (const float*)d_in[7];
  float* out = (float*)d_out;

  _Float16* W2h = (_Float16*)d_ws;                                        // 36*2048
  _Float16* W3h = (_Float16*)((char*)d_ws + 36 * 2048 * sizeof(_Float16)); // 36*7168

  pack_weights<<<36, 256, 0, stream>>>(W2, W3, W2h, W3h);

  const int smemBytes = 145792;
  (void)hipFuncSetAttribute((const void*)se_fused,
                            hipFuncAttributeMaxDynamicSharedMemorySize, smemBytes);
  dim3 grid(NCAT, NBATCH / MB);
  se_fused<<<grid, 128, smemBytes, stream>>>(coords, typesP, W1, b1, b2, b3,
                                             W2h, W3h, out);
}
